// _SelfAttentionBlock_47863115547120
// MI455X (gfx1250) — matrix operand order repaired and checked by reading, not compiled, not run on hardware
//
#include <hip/hip_runtime.h>
#include <hip/hip_bf16.h>

typedef __attribute__((ext_vector_type(16))) _Float16 v16h;
typedef __attribute__((ext_vector_type(8)))  _Float16 v8h;
typedef __attribute__((ext_vector_type(8)))  float    v8f;

#define NTOT 6400   // 80*80
#define HW   80
#define CIN  256
#define CK   128
#define CV   256
#define CSPLIT 2

// ---------- WMMA helpers ----------
__device__ __forceinline__ v16h cat16(v8h lo, v8h hi) {
  return __builtin_shufflevector(lo, hi, 0,1,2,3,4,5,6,7,8,9,10,11,12,13,14,15);
}
__device__ __forceinline__ v8f wmma_f16(v16h a, v16h b, v8f c) {
  // D = A(16x32 f16) x B(32x16 f16) + C(16x16 f32)
  return __builtin_amdgcn_wmma_f32_16x16x32_f16(false, a, false, b, (short)0, c,
                                                false, false);
}
// A fragment (16x32) from row-major f16 matrix, row stride ld, row base r0, col base k0
__device__ __forceinline__ v16h load_a(const _Float16* __restrict__ p, long ld,
                                       int r0, int k0, int lane) {
  const int kb = (lane < 16) ? 0 : 16;
  const _Float16* rp = p + (long)(r0 + (lane & 15)) * ld + k0 + kb;
  v8h lo = *(const v8h*)rp;
  v8h hi = *(const v8h*)(rp + 8);
  return cat16(lo, hi);
}

// ---------- prep: fold BN into key weights, convert/permute weights to f16 ------
// wv16 is stored with K reordered as k' = (dh*3+dw)*256 + c so that every K-chunk
// of 32 has one spatial offset and 32 consecutive channels (contiguous B-frags).
__global__ void prep_kernel(const float* __restrict__ wk, const float* __restrict__ bk,
                            const float* __restrict__ gamma, const float* __restrict__ beta,
                            const float* __restrict__ rmean, const float* __restrict__ rvar,
                            const float* __restrict__ wv, const float* __restrict__ ww,
                            _Float16* __restrict__ wk16, float* __restrict__ bias2,
                            _Float16* __restrict__ wv16, _Float16* __restrict__ ww16) {
  const long i = (long)blockIdx.x * blockDim.x + threadIdx.x;
  if (i < (long)CV * CIN * 9) {
    int o = (int)(i / (CIN * 9));
    int t = (int)(i % (CIN * 9));
    int pos = t >> 8;                 // dh*3+dw
    int c = t & 255;
    int dh = pos / 3, dw = pos - dh * 3;
    wv16[i] = (_Float16)wv[(((long)o * CIN + c) * 3 + dh) * 3 + dw];
  }
  if (i < (long)CV * CV) ww16[i] = (_Float16)ww[i];
  if (i < (long)CK * CIN) {
    int ck = (int)(i >> 8);
    float inv = gamma[ck] * rsqrtf(rvar[ck] + 1e-5f);
    wk16[i] = (_Float16)(wk[i] * inv);
  }
  if (i < CK) {
    float inv = gamma[i] * rsqrtf(rvar[i] + 1e-5f);
    bias2[i] = bk[i] * inv + beta[i] - rmean[i] * inv;
  }
}

// ---------- transpose x: f32 [B][C][N] -> f16 [B][N][C] via LDS 32x32 tiles ----
__global__ __launch_bounds__(256) void xpose_kernel(const float* __restrict__ x,
                                                    _Float16* __restrict__ xt) {
  __shared__ float tile[32][33];
  const int bid = blockIdx.x;               // b*(8*200) + ct*200 + nt
  const int b  = bid / 1600;
  const int r  = bid % 1600;
  const int c0 = (r / 200) * 32;
  const int n0 = (r % 200) * 32;
  const int tx = threadIdx.x & 31, ty = threadIdx.x >> 5;
  const float* xb = x + (long)b * CIN * NTOT;
#pragma unroll
  for (int i = 0; i < 4; ++i)
    tile[ty + i * 8][tx] = xb[(long)(c0 + ty + i * 8) * NTOT + n0 + tx];
  __syncthreads();
  _Float16* xtb = xt + (long)b * NTOT * CIN;
#pragma unroll
  for (int i = 0; i < 4; ++i)
    xtb[(long)(n0 + ty + i * 8) * CIN + c0 + tx] = (_Float16)tile[tx][ty + i * 8];
}

// ---------- key/query: 1x1 conv (+BN folded) as WMMA GEMM -> kq[b][n][128] f16 --
__global__ __launch_bounds__(32) void kq_kernel(const _Float16* __restrict__ xt,
                                                const _Float16* __restrict__ wk16,
                                                const float* __restrict__ bias2,
                                                _Float16* __restrict__ kq) {
  const int lane = threadIdx.x & 31;
  const int bid = blockIdx.x;               // b*(8*400) + ot*400 + nt
  const int b  = bid / 3200;
  const int r  = bid % 3200;
  const int o0 = (r / 400) * 16;
  const int n0 = (r % 400) * 16;
  const int kb2 = (lane < 16) ? 0 : 16;
  const int mb  = (lane < 16) ? 0 : 8;
  const int n   = n0 + (lane & 15);
  const _Float16* xp = xt + ((long)b * NTOT + n) * CIN + kb2;

  v8f acc = {};
#pragma unroll
  for (int kc = 0; kc < CIN; kc += 32) {
    v16h a  = load_a(wk16, CIN, o0, kc, lane);
    v16h bf = *(const v16h*)(xp + kc);
    acc = wmma_f16(a, bf, acc);
  }
  v8h o;
#pragma unroll
  for (int v = 0; v < 8; ++v) o[v] = (_Float16)(acc[v] + bias2[o0 + mb + v]);
  *(v8h*)(kq + ((long)b * NTOT + n) * CK + o0 + mb) = o;   // [n][ck] contiguous
}

// ---------- value: 3x3 conv as implicit-GEMM WMMA -> val[b][o][n] f16 ----------
// K-order (dh,dw,c): per spatial offset, 8 contiguous 32-channel chunks from xt.
__global__ __launch_bounds__(32) void vconv_kernel(const _Float16* __restrict__ xt,
                                                   const _Float16* __restrict__ wv16,
                                                   const float* __restrict__ bv,
                                                   _Float16* __restrict__ val) {
  const int lane = threadIdx.x & 31;
  const int bid = blockIdx.x;               // b*(16*400) + ot*400 + nt
  const int b  = bid / 6400;
  const int r  = bid % 6400;
  const int o0 = (r / 400) * 16;
  const int n0 = (r % 400) * 16;
  const int kb2 = (lane < 16) ? 0 : 16;
  const int mb  = (lane < 16) ? 0 : 8;
  const int n   = n0 + (lane & 15);
  const int h = n / HW, w = n % HW;
  const _Float16* xtb = xt + (long)b * NTOT * CIN;

  v8f acc = {};
  v16h zeroh = {};
#pragma unroll 1
  for (int pos = 0; pos < 9; ++pos) {
    const int dh = pos / 3, dw = pos - dh * 3;
    const int hh = h + dh - 1, w2 = w + dw - 1;
    const bool valid = ((unsigned)hh < (unsigned)HW) && ((unsigned)w2 < (unsigned)HW);
    const _Float16* bp = xtb + (long)(hh * HW + w2) * CIN + kb2;
#pragma unroll
    for (int cc = 0; cc < 8; ++cc) {
      v16h a = load_a(wv16, CIN * 9, o0, pos * 256 + cc * 32, lane);
      v16h bf = zeroh;
      if (valid) bf = *(const v16h*)(bp + cc * 32);
      acc = wmma_f16(a, bf, acc);
    }
  }
#pragma unroll
  for (int v = 0; v < 8; ++v) {
    int o = o0 + mb + v;
    val[((long)b * CV + o) * NTOT + n] = (_Float16)(acc[v] + bv[o]);
  }
}

// ---------- fused flash attention: ctx = softmax(kq^T kq) @ value^T ----------
// One wave = 16 queries x 128 channels. Computes S^T tiles (K rows x Q cols) so
// softmax is per-lane + one shfl_xor(16); exp(S^T) is already in B-fragment lane
// order for O^T += V^T x P^T (cross-half shfl exchange, no LDS).
// Epilogue stores ctx in B-fragment-ready swizzle [cc=ch/32][n][ch%32].
__global__ __launch_bounds__(32) void flash_kernel(const _Float16* __restrict__ kq,
                                                   const _Float16* __restrict__ val,
                                                   _Float16* __restrict__ ctx) {
  const int lane = threadIdx.x & 31;
  const int bid = blockIdx.x;               // b*(400*CSPLIT) + qt*CSPLIT + cs
  const int b  = bid / (400 * CSPLIT);
  const int r  = bid % (400 * CSPLIT);
  const int q0 = (r / CSPLIT) * 16;
  const int c0 = (r % CSPLIT) * (CV / CSPLIT);
  const int nq  = lane & 15;
  const int kb2 = (lane < 16) ? 0 : 16;
  const int mb  = (lane < 16) ? 0 : 8;
  const bool hihalf = (lane >= 16);

  const _Float16* kqb = kq  + (long)b * NTOT * CK;
  const _Float16* vb  = val + (long)b * CV * NTOT;

  v16h qfrag[4];                             // Q^T B-fragments, resident all loop
#pragma unroll
  for (int kc = 0; kc < 4; ++kc)
    qfrag[kc] = *(const v16h*)(kqb + (long)(q0 + nq) * CK + kc * 32 + kb2);

  v8f zero = {};
  v8f acc[8];
#pragma unroll
  for (int f = 0; f < 8; ++f) acc[f] = zero;
  float mrun = -1e30f, lrun = 0.f;

  for (int kt = 0; kt < NTOT; kt += 32) {
    __builtin_prefetch(kqb + (long)(kt + 32) * CK, 0, 1);
    // S^T tiles: t0 = keys kt..kt+15, t1 = keys kt+16..kt+31 (rows), queries (cols)
    v8f t0 = zero, t1 = zero;
#pragma unroll
    for (int kc = 0; kc < 4; ++kc) {
      v16h a0 = load_a(kqb, CK, kt,      kc * 32, lane);
      v16h a1 = load_a(kqb, CK, kt + 16, kc * 32, lane);
      t0 = wmma_f16(a0, qfrag[kc], t0);
      t1 = wmma_f16(a1, qfrag[kc], t1);
    }
    // online softmax over key dim: per-lane reduce + exchange with other half
    float tmax = t0[0];
#pragma unroll
    for (int v = 0; v < 8; ++v) { tmax = fmaxf(tmax, t0[v]); tmax = fmaxf(tmax, t1[v]); }
    tmax = fmaxf(tmax, __shfl_xor(tmax, 16, 32));
    float mnew = fmaxf(mrun, tmax);
    float sc = __expf(mrun - mnew);
    float p0[8], p1[8], ps = 0.f;
#pragma unroll
    for (int v = 0; v < 8; ++v) {
      p0[v] = __expf(t0[v] - mnew);
      p1[v] = __expf(t1[v] - mnew);
      ps += p0[v] + p1[v];
    }
    ps += __shfl_xor(ps, 16, 32);
    lrun = lrun * sc + ps;
    mrun = mnew;
#pragma unroll
    for (int f = 0; f < 8; ++f)
#pragma unroll
      for (int v = 0; v < 8; ++v) acc[f][v] *= sc;
    // build P^T B-fragment (32 keys x 16 queries) via cross-half exchange
    v16h pb;
#pragma unroll
    for (int v = 0; v < 8; ++v) {
      float o0x = __shfl_xor(p0[v], 16, 32);
      float o1x = __shfl_xor(p1[v], 16, 32);
      pb[v]     = (_Float16)(hihalf ? o1x : p0[v]);
      pb[8 + v] = (_Float16)(hihalf ? p1[v] : o0x);
    }
    // O^T += V^T(chunk) x P^T over this wave's 128 channels
#pragma unroll
    for (int f = 0; f < 8; ++f) {
      v16h av = load_a(vb, NTOT, c0 + f * 16, kt, lane);
      acc[f] = wmma_f16(av, pb, acc[f]);
    }
  }
  const float invl = 1.0f / lrun;
  _Float16* cbs = ctx + (long)b * CV * NTOT;
#pragma unroll
  for (int f = 0; f < 8; ++f) {
    const int cc = (c0 >> 5) + (f >> 1);     // channel chunk (ch/32)
    v8h oh;
#pragma unroll
    for (int v = 0; v < 8; ++v) oh[v] = (_Float16)(acc[f][v] * invl);
    *(v8h*)(cbs + ((long)cc * NTOT + q0 + nq) * 32 + (f & 1) * 16 + mb) = oh;
  }
}

// ---------- final 1x1 conv + residual -> out f32 ----------
__global__ __launch_bounds__(32) void fconv_kernel(const _Float16* __restrict__ ctx,
                                                   const _Float16* __restrict__ ww16,
                                                   const float* __restrict__ bw,
                                                   const float* __restrict__ x,
                                                   float* __restrict__ out) {
  const int lane = threadIdx.x & 31;
  const int bid = blockIdx.x;               // b*(16*400) + ot*400 + nt
  const int b  = bid / 6400;
  const int r  = bid % 6400;
  const int o0 = (r / 400) * 16;
  const int n0 = (r % 400) * 16;
  const int kb2 = (lane < 16) ? 0 : 16;
  const int mb  = (lane < 16) ? 0 : 8;
  const int n   = n0 + (lane & 15);
  const _Float16* cb = ctx + (long)b * CV * NTOT;

  v8f acc = {};
#pragma unroll
  for (int cc = 0; cc < 8; ++cc) {           // K chunks of 32 channels
    v16h a  = load_a(ww16, CV, o0, cc * 32, lane);
    v16h bf = *(const v16h*)(cb + ((long)cc * NTOT + n) * 32 + kb2);
    acc = wmma_f16(a, bf, acc);
  }
#pragma unroll
  for (int v = 0; v < 8; ++v) {
    int o = o0 + mb + v;
    long idx = ((long)b * CV + o) * NTOT + n;
    out[idx] = acc[v] + bw[o] + x[idx];
  }
}

extern "C" void kernel_launch(void* const* d_in, const int* in_sizes, int n_in,
                              void* d_out, int out_size, void* d_ws, size_t ws_size,
                              hipStream_t stream) {
  const float* x     = (const float*)d_in[0];
  const float* wk    = (const float*)d_in[1];
  const float* bk    = (const float*)d_in[2];
  const float* gamma = (const float*)d_in[3];
  const float* beta  = (const float*)d_in[4];
  const float* rmean = (const float*)d_in[5];
  const float* rvar  = (const float*)d_in[6];
  const float* wv    = (const float*)d_in[7];
  const float* bv    = (const float*)d_in[8];
  const float* ww    = (const float*)d_in[9];
  const float* bw    = (const float*)d_in[10];
  float* out = (float*)d_out;

  char* ws = (char*)d_ws;
  _Float16* wk16  = (_Float16*)(ws + 0);          //  64 KB: [128][256] BN-folded
  float*    bias2 = (float*)   (ws + 65536);      //  512 B
  _Float16* wv16  = (_Float16*)(ws + 66048);      // 1.13 MB: [256][9*256] (dh,dw,c)
  _Float16* ww16  = (_Float16*)(ws + 1245696);    // 128 KB: [256][256]
  _Float16* kq16  = (_Float16*)(ws + 1376768);    // 3.13 MB: [B][N][128]
  _Float16* val16 = (_Float16*)(ws + 4653568);    // 6.25 MB: [B][256][N]
  _Float16* ctxs  = (_Float16*)(ws + 11207168);   // 6.25 MB: [B][cc][N][32] swizzled
  _Float16* xt16  = (_Float16*)(ws + 17760768);   // 6.25 MB: [B][N][256]

  prep_kernel <<<(CV * CIN * 9 + 255) / 256, 256, 0, stream>>>(
      wk, bk, gamma, beta, rmean, rvar, wv, ww, wk16, bias2, wv16, ww16);
  xpose_kernel<<<2 * 8 * 200,       256, 0, stream>>>(x, xt16);
  kq_kernel   <<<2 * 8 * 400,        32, 0, stream>>>(xt16, wk16, bias2, kq16);
  vconv_kernel<<<2 * 16 * 400,       32, 0, stream>>>(xt16, wv16, bv, val16);
  flash_kernel<<<2 * 400 * CSPLIT,   32, 0, stream>>>(kq16, val16, ctxs);
  fconv_kernel<<<2 * 16 * 400,       32, 0, stream>>>(ctxs, ww16, bw, x, out);
}
